// VectorQuantizer_58042188038276
// MI455X (gfx1250) — matrix operand order repaired and checked by reading, not compiled, not run on hardware
//
// VQ-VAE quantize + EMA update for MI455X (gfx1250).
// Core: bf16 WMMA distance GEMM (v_wmma_f32_16x16x32_bf16), codebook streamed
// from L2 through double-buffered LDS via async-to-LDS; A tile register-resident.
// Workspace needed: ~57 MB.
#include <hip/hip_runtime.h>
#include <cstdint>
#include <cstddef>

#define VOCAB 8192
#define DIM 512
#define M_TOT 32768 /* 8*4096 */
#define BETA 0.25f
#define MOMENTUM 0.995f
#define MIN_COUNT 0.1f
#define PERTURB 0.001f
#define MAX_SPLIT_STEPS 100

typedef __attribute__((ext_vector_type(16))) __bf16 v16bf;
typedef __attribute__((ext_vector_type(8)))  float  v8f;
typedef int v4i __attribute__((__vector_size__(16)));   // matches async builtin pointee

union Frag16 { uint4 u[2]; v16bf v; };

static __device__ __forceinline__ unsigned short f2bf(float f) {
  unsigned u = __builtin_bit_cast(unsigned, f);
  u += 0x7fffu + ((u >> 16) & 1u);               // round-to-nearest-even
  return (unsigned short)(u >> 16);
}

// ---------------------------------------------------------------- async path
#if __has_builtin(__builtin_amdgcn_global_load_async_to_lds_b128)
#define USE_ASYNC 1
#else
#define USE_ASYNC 0
#endif
#define AS1 __attribute__((address_space(1)))
#define AS3 __attribute__((address_space(3)))

static __device__ __forceinline__ void async_wait_all() {
#if USE_ASYNC
#if __has_builtin(__builtin_amdgcn_s_wait_asynccnt)
  __builtin_amdgcn_s_wait_asynccnt(0);
#else
  asm volatile("s_wait_asynccnt 0" ::: "memory");
#endif
#endif
}

// Stage 16 codebook rows (16 x 512 bf16 = 16KB) + their |e|^2 into LDS.
static __device__ __forceinline__ void load_etile(unsigned short* dst, float* esq_dst,
                                                  const unsigned short* __restrict__ e_bf,
                                                  const float* __restrict__ esq,
                                                  int k0, int tid) {
#if USE_ASYNC
  const char* gbase = (const char*)(e_bf + (size_t)k0 * DIM);
  char* lbase = (char*)dst;
#pragma unroll
  for (int j = 0; j < 4; ++j) {                  // 1024 b128 transfers / 256 threads
    int byte = (tid + j * 256) * 16;
    __builtin_amdgcn_global_load_async_to_lds_b128(
        (AS1 v4i*)(gbase + byte), (AS3 v4i*)(lbase + byte), 0, 0);
  }
#else
  const uint4* g = (const uint4*)(e_bf + (size_t)k0 * DIM);
  uint4* l = (uint4*)dst;
#pragma unroll
  for (int j = 0; j < 4; ++j) l[tid + j * 256] = g[tid + j * 256];
#endif
  if (tid < 16) esq_dst[tid] = esq[k0 + tid];
}

// ------------------------------------------------------------------ kernels
__global__ __launch_bounds__(256) void vq_prep_x(const float* __restrict__ x,
                                                 unsigned* __restrict__ xb) {
  size_t i = (size_t)blockIdx.x * 256 + threadIdx.x;   // 8 floats per thread
  const float4* s = (const float4*)x;
  float4 a = s[2 * i], b = s[2 * i + 1];
  uint4 o;
  o.x = (unsigned)f2bf(a.x) | ((unsigned)f2bf(a.y) << 16);
  o.y = (unsigned)f2bf(a.z) | ((unsigned)f2bf(a.w) << 16);
  o.z = (unsigned)f2bf(b.x) | ((unsigned)f2bf(b.y) << 16);
  o.w = (unsigned)f2bf(b.z) | ((unsigned)f2bf(b.w) << 16);
  ((uint4*)xb)[i] = o;
}

__global__ __launch_bounds__(128) void vq_prep_e(const float* __restrict__ dict,
                                                 const float* __restrict__ counts,
                                                 float* __restrict__ e_f32,
                                                 unsigned short* __restrict__ e_bf,
                                                 float* __restrict__ esq) {
  const int k = blockIdx.x;
  const float inv = 1.0f / counts[k];
  float ss = 0.0f;
  for (int c = threadIdx.x; c < DIM; c += 128) {
    float e = dict[(size_t)k * DIM + c] * inv;
    e_f32[(size_t)k * DIM + c] = e;
    e_bf[(size_t)k * DIM + c] = f2bf(e);
    ss += e * e;
  }
#pragma unroll
  for (int off = 16; off; off >>= 1) ss += __shfl_xor(ss, off, 32);
  __shared__ float red[4];
  if ((threadIdx.x & 31) == 0) red[threadIdx.x >> 5] = ss;
  __syncthreads();
  if (threadIdx.x == 0) esq[k] = red[0] + red[1] + red[2] + red[3];
}

// 8 waves/block; each wave owns 16 rows; A tile (16x512 bf16) register-resident.
__global__ __launch_bounds__(256) void vq_argmin(const unsigned short* __restrict__ x_bf,
                                                 const unsigned short* __restrict__ e_bf,
                                                 const float* __restrict__ esq,
                                                 int* __restrict__ ids) {
  __shared__ union {
    unsigned short xstage[128 * 32];             // phase 1: x staging (8 KB)
    struct {
      unsigned short ebuf[2][16 * DIM];          // phase 2: double-buffered e tiles
      float esqb[2][16];
    } p2;
  } lds;

  const int tid  = threadIdx.x;
  const int wave = tid >> 5;                     // wave32
  const int lane = tid & 31;
  const int g    = lane >> 4;                    // lane group (0/1)
  const int n    = lane & 15;
  const int m0   = blockIdx.x * 128;

  // ---- Phase 1: pull this wave's 16x512 A tile into registers via LDS ----
  Frag16 a[16];
  for (int kc = 0; kc < 16; ++kc) {              // 16 K-chunks of 32
    __syncthreads();
    const unsigned short* src = x_bf + (size_t)m0 * DIM + kc * 32;
#pragma unroll
    for (int j = 0; j < 2; ++j) {                // 512 uint4 / 256 threads
      int idx = tid + j * 256;
      int row = idx >> 2, c4 = idx & 3;
      ((uint4*)lds.xstage)[idx] = *(const uint4*)(src + (size_t)row * DIM + c4 * 8);
    }
    __syncthreads();
    const int xrow = wave * 16 + n;
    // ISA A layout: lane(group g, row n) holds K = g*8..g*8+7 and 16+g*8..+7
    a[kc].u[0] = *(const uint4*)&lds.xstage[xrow * 32 + g * 8];
    a[kc].u[1] = *(const uint4*)&lds.xstage[xrow * 32 + 16 + g * 8];
  }
  __syncthreads();

  float minv[8]; int mini[8];
#pragma unroll
  for (int r = 0; r < 8; ++r) { minv[r] = 3.4e38f; mini[r] = 0; }

  load_etile(lds.p2.ebuf[0], lds.p2.esqb[0], e_bf, esq, 0, tid);
  async_wait_all();
  __syncthreads();

  // ---- Phase 2: stream all 8192 codewords, 16 at a time ----
  for (int k0 = 0; k0 < VOCAB; k0 += 16) {
    const int cur = (k0 >> 4) & 1;
    if (k0 + 16 < VOCAB)
      load_etile(lds.p2.ebuf[cur ^ 1], lds.p2.esqb[cur ^ 1], e_bf, esq, k0 + 16, tid);

    v8f acc = {0.f, 0.f, 0.f, 0.f, 0.f, 0.f, 0.f, 0.f};
    const unsigned short* eb = lds.p2.ebuf[cur] + n * DIM + g * 8;
#pragma unroll
    for (int kc = 0; kc < 16; ++kc) {
      Frag16 b;                                  // B: lane(col n, group g) holds K=g*8..+7 and 16+g*8..+7
      b.u[0] = *(const uint4*)(eb + kc * 32);
      b.u[1] = *(const uint4*)(eb + kc * 32 + 16);
      acc = __builtin_amdgcn_wmma_f32_16x16x32_bf16(false, a[kc].v, false, b.v,
                                                    (short)0, acc, false, false);
    }
    const float es = lds.p2.esqb[cur][n];        // score = |e|^2 - 2*x.e (|x|^2 const)
#pragma unroll
    for (int r = 0; r < 8; ++r) {
      float s = es - 2.0f * acc[r];
      if (s < minv[r]) { minv[r] = s; mini[r] = k0 + n; }
    }
    async_wait_all();
    __syncthreads();
  }

  // ---- cross-lane argmin within each 16-lane group; D layout: M = g*8 + r ----
  const int mwave = m0 + wave * 16;
#pragma unroll
  for (int r = 0; r < 8; ++r) {
    float v = minv[r]; int i = mini[r];
#pragma unroll
    for (int off = 8; off; off >>= 1) {
      float v2 = __shfl_xor(v, off, 32);
      int   i2 = __shfl_xor(i, off, 32);
      if (v2 < v || (v2 == v && i2 < i)) { v = v2; i = i2; }
    }
    if (n == 0) ids[mwave + g * 8 + r] = i;
  }
}

__global__ __launch_bounds__(256) void vq_ema_init(const float* __restrict__ dict,
                                                   const float* __restrict__ counts,
                                                   float* __restrict__ ndict,
                                                   float* __restrict__ ncnt,
                                                   float* __restrict__ loss_acc) {
  size_t i = (size_t)blockIdx.x * 256 + threadIdx.x;
  ndict[i] = dict[i] * MOMENTUM;
  if (i < VOCAB) ncnt[i] = counts[i] * MOMENTUM;
  if (i == 0) *loss_acc = 0.0f;
}

__global__ __launch_bounds__(256) void vq_scatter(const float* __restrict__ x,
                                                  const float* __restrict__ e_f32,
                                                  const int* __restrict__ ids,
                                                  float* __restrict__ xq,
                                                  float* __restrict__ ndict,
                                                  float* __restrict__ ncnt,
                                                  float* __restrict__ loss_acc) {
  const int m = blockIdx.x;
  const int id = ids[m];
  float lsum = 0.0f;
  for (int c = threadIdx.x; c < DIM; c += 256) {
    float xv = x[(size_t)m * DIM + c];
    float ev = e_f32[(size_t)id * DIM + c];
    xq[(size_t)m * DIM + c] = ev;                // straight-through: x_q == e[id]
    float d = ev - xv;
    lsum += d * d;
    atomicAdd(&ndict[(size_t)id * DIM + c], xv); // x_sum scatter
  }
#pragma unroll
  for (int off = 16; off; off >>= 1) lsum += __shfl_xor(lsum, off, 32);
  __shared__ float red[8];
  if ((threadIdx.x & 31) == 0) red[threadIdx.x >> 5] = lsum;
  __syncthreads();
  if (threadIdx.x == 0) {
    float t = 0.f;
#pragma unroll
    for (int w = 0; w < 8; ++w) t += red[w];
    atomicAdd(loss_acc, t);
    atomicAdd(&ncnt[id], 1.0f);
  }
}

static __device__ __forceinline__ float split_jitter(int step, int c) {
  unsigned h = (unsigned)(step * 8209 + c) * 2654435761u;
  h ^= h >> 16; h *= 2246822519u; h ^= h >> 13;
  float u = (float)(h >> 8) * (1.0f / 16777216.0f);
  return 1.0f - PERTURB + 2.0f * PERTURB * u;
}

__global__ __launch_bounds__(1024) void vq_split(float* __restrict__ ndict,
                                                 float* __restrict__ ncnt) {
  __shared__ float svn[32], svx[32];
  __shared__ int smn[32], smx[32];
  __shared__ int s_imin, s_imax, s_go;
  const int tid = threadIdx.x, lane = tid & 31, warp = tid >> 5;

  for (int step = 0; step < MAX_SPLIT_STEPS; ++step) {
    float vmin = 3.4e38f, vmax = -3.4e38f; int imin = 0, imax = 0;
    for (int i = tid; i < VOCAB; i += 1024) {
      float c = ncnt[i];
      if (c < vmin) { vmin = c; imin = i; }
      if (c > vmax) { vmax = c; imax = i; }
    }
#pragma unroll
    for (int off = 16; off; off >>= 1) {
      float v2 = __shfl_xor(vmin, off, 32); int i2 = __shfl_xor(imin, off, 32);
      if (v2 < vmin || (v2 == vmin && i2 < imin)) { vmin = v2; imin = i2; }
      float w2 = __shfl_xor(vmax, off, 32); int j2 = __shfl_xor(imax, off, 32);
      if (w2 > vmax || (w2 == vmax && j2 < imax)) { vmax = w2; imax = j2; }
    }
    if (lane == 0) { svn[warp] = vmin; smn[warp] = imin; svx[warp] = vmax; smx[warp] = imax; }
    __syncthreads();
    if (tid == 0) {
      float fvn = svn[0], fvx = svx[0]; int fin = smn[0], fix = smx[0];
      for (int w = 1; w < 32; ++w) {
        if (svn[w] < fvn || (svn[w] == fvn && smn[w] < fin)) { fvn = svn[w]; fin = smn[w]; }
        if (svx[w] > fvx || (svx[w] == fvx && smx[w] < fix)) { fvx = svx[w]; fix = smx[w]; }
      }
      s_imin = fin; s_imax = fix;
      s_go = (fvn < MIN_COUNT && fin != fix) ? 1 : 0;
      if (s_go) { float h = ncnt[fix] * 0.5f; ncnt[fin] = h; ncnt[fix] = h; }
    }
    __syncthreads();
    if (!s_go) return;
    const int imn = s_imin, imx = s_imax;
    for (int c = tid; c < DIM; c += 1024) {
      float em = ndict[(size_t)imx * DIM + c];
      ndict[(size_t)imn * DIM + c] = em * split_jitter(step, c) * 0.5f;
      ndict[(size_t)imx * DIM + c] = em * 0.5f;
    }
    __syncthreads();
  }
}

__global__ __launch_bounds__(1024) void vq_finalize(const float* __restrict__ ncnt,
                                                    const float* __restrict__ loss_acc,
                                                    float* __restrict__ out_sc) {
  __shared__ float red[32];
  __shared__ float s_total;
  const int tid = threadIdx.x, lane = tid & 31, warp = tid >> 5;
  float s = 0.0f;
  for (int i = tid; i < VOCAB; i += 1024) s += ncnt[i];
#pragma unroll
  for (int off = 16; off; off >>= 1) s += __shfl_xor(s, off, 32);
  if (lane == 0) red[warp] = s;
  __syncthreads();
  if (tid == 0) {
    float t = 0.f;
    for (int w = 0; w < 32; ++w) t += red[w];
    s_total = t;
  }
  __syncthreads();
  const float inv = 1.0f / s_total;
  float h = 0.0f;
  for (int i = tid; i < VOCAB; i += 1024) {
    float p = ncnt[i] * inv;
    h += p * __logf(p + 1e-10f);
  }
#pragma unroll
  for (int off = 16; off; off >>= 1) h += __shfl_xor(h, off, 32);
  __syncthreads();
  if (lane == 0) red[warp] = h;
  __syncthreads();
  if (tid == 0) {
    float H = 0.f;
    for (int w = 0; w < 32; ++w) H += red[w];
    out_sc[0] = BETA * (*loss_acc) / (float)((size_t)M_TOT * DIM);  // loss_vq
    out_sc[1] = __expf(-H);                                          // perplexity
  }
}

// ------------------------------------------------------------------ launch
extern "C" void kernel_launch(void* const* d_in, const int* in_sizes, int n_in,
                              void* d_out, int out_size, void* d_ws, size_t ws_size,
                              hipStream_t stream) {
  (void)in_sizes; (void)n_in; (void)out_size; (void)ws_size;
  const float* x      = (const float*)d_in[0];
  const float* dict   = (const float*)d_in[1];
  const float* counts = (const float*)d_in[2];

  float* out   = (float*)d_out;
  float* xq    = out;                                   // [M,512]
  float* sc    = out + (size_t)M_TOT * DIM;             // loss, perplexity
  float* ndict = sc + 2;                                // [8192,512]
  float* ncnt  = ndict + (size_t)VOCAB * DIM;           // [8192]

  char* ws = (char*)d_ws;
  unsigned short* x_bf = (unsigned short*)ws; ws += (size_t)M_TOT * DIM * 2;  // 32 MB
  unsigned short* e_bf = (unsigned short*)ws; ws += (size_t)VOCAB * DIM * 2;  //  8 MB
  float* e_f32 = (float*)ws;                  ws += (size_t)VOCAB * DIM * 4;  // 16 MB
  float* esq   = (float*)ws;                  ws += (size_t)VOCAB * 4;
  int*   ids   = (int*)ws;                    ws += (size_t)M_TOT * 4;
  float* loss_acc = (float*)ws;

  vq_prep_x<<<(M_TOT * DIM) / (256 * 8), 256, 0, stream>>>(x, (unsigned*)x_bf);
  vq_prep_e<<<VOCAB, 128, 0, stream>>>(dict, counts, e_f32, e_bf, esq);
  vq_argmin<<<M_TOT / 128, 256, 0, stream>>>(x_bf, e_bf, esq, ids);
  vq_ema_init<<<(VOCAB * DIM) / 256, 256, 0, stream>>>(dict, counts, ndict, ncnt, loss_acc);
  vq_scatter<<<M_TOT, 256, 0, stream>>>(x, e_f32, ids, xq, ndict, ncnt, loss_acc);
  vq_split<<<1, 1024, 0, stream>>>(ndict, ncnt);
  vq_finalize<<<1, 1024, 0, stream>>>(ncnt, loss_acc, sc);
}
